// MarianAttention_90709709291977
// MI455X (gfx1250) — compile-verified, matrix operand order checked by reading, not run on hardware
//
#include <hip/hip_runtime.h>

#define DEVFN __device__ __forceinline__

typedef __bf16 bf16_t;
typedef __attribute__((ext_vector_type(16))) __bf16 v16bf;
typedef __attribute__((ext_vector_type(8)))  float  v8f;
typedef __attribute__((ext_vector_type(4)))  float  v4f;
typedef __attribute__((ext_vector_type(4)))  __bf16 v4bf;
typedef unsigned int u32;
typedef u32 u32x4 __attribute__((ext_vector_type(4)));
typedef u32 u32x8 __attribute__((ext_vector_type(8)));

constexpr int B = 2, T = 2048, D = 1024, H = 16, DH = 64;
constexpr float EPS = 1e-7f;
constexpr float HIV = 1.0f - 1e-7f;

DEVFN float clampf(float x) { x = (x < EPS) ? EPS : x; return (x > HIV) ? HIV : x; }
DEVFN float clamp_nan(float x) { return __builtin_isnan(x) ? 0.0f : clampf(x); }

union FragU { v16bf v; uint4 q[2]; };

// 16x32 bf16 WMMA fragment load. Lane l handles row (l&15); lanes 0-15 take K
// chunks {0..7,16..23}, lanes 16-31 take {8..15,24..31} (CDNA5 16-bit layout).
DEVFN v16bf load_frag(const bf16_t* __restrict__ base, int stride) {
  const int lane = threadIdx.x & 31;
  const bf16_t* p = base + (lane & 15) * stride + ((lane >> 4) << 3);
  FragU f;
  f.q[0] = *reinterpret_cast<const uint4*>(p);
  f.q[1] = *reinterpret_cast<const uint4*>(p + 16);
  return f.v;
}

DEVFN v8f wmma_bf16(v16bf a, v16bf b, v8f c) {
  return __builtin_amdgcn_wmma_f32_16x16x32_bf16(
      false, a, false, b, (short)0, c, false, false);
}

// ---- CDNA5 async memory->LDS copy (ASYNCcnt) --------------------------------
DEVFN void async_ld_b128(unsigned lds_off, unsigned long long gaddr) {
  asm volatile("global_load_async_to_lds_b128 %0, %1, off"
               :: "v"(lds_off), "v"(gaddr) : "memory");
}
DEVFN void wait_async_le4(void) { asm volatile("s_wait_asynccnt 0x4" ::: "memory"); }
DEVFN void wait_async_0(void)   { asm volatile("s_wait_asynccnt 0x0" ::: "memory"); }

// ---- CDNA5 Tensor Data Mover: 2D bf16 tile -> LDS (TENSORcnt) ---------------
// D# group0: count=1 | lds_addr | global_addr[56:0] | type=2
// D# group1: data_size=2B, tile 64x64, tensor dims 64x64, dim0 stride 64 elems
DEVFN void tdm_load_k64x64(u32 lds_addr, const bf16_t* gsrc) {
  const unsigned long long ga = (unsigned long long)(size_t)gsrc;
  u32x4 g0;
  g0[0] = 1u;                                            // count=1, user desc
  g0[1] = lds_addr;                                      // LDS byte address
  g0[2] = (u32)ga;                                       // global_addr[31:0]
  g0[3] = (u32)((ga >> 32) & 0x01FFFFFFu) | (2u << 30);  // [56:32] | type=2
  u32x8 g1;
  g1[0] = 1u << 16;            // workgroup_mask=0, data_size=1 (2 bytes)
  g1[1] = 64u << 16;           // tensor_dim0[15:0]=64 (atomic_barrier_addr=0)
  g1[2] = 64u << 16;           // tensor_dim0 hi=0, tensor_dim1[15:0]=64
  g1[3] = 64u << 16;           // tensor_dim1 hi=0, tile_dim0=64
  g1[4] = 64u;                 // tile_dim1=64, tile_dim2=0
  g1[5] = 64u;                 // tensor_dim0_stride = 64 elements
  g1[6] = 0u;                  // stride hi / dim1_stride (unused, 2D tile)
  g1[7] = 0u;
  asm volatile("tensor_load_to_lds %0, %1" :: "s"(g0), "s"(g1) : "memory");
}

// Stage a 64-row x 64-bf16 (8 KB) V tile into LDS with async copies.
// 512 x 16B chunks, 128 threads -> 4 async ops per thread. Row stride = T.
DEVFN void stage_v_async(const bf16_t* gsrc, unsigned lds_base) {
  const int tid = threadIdx.x;  // 0..127
#pragma unroll
  for (int j = 0; j < 4; ++j) {
    const int c = tid + j * 128;          // chunk 0..511
    const int row = c >> 3;               // 64 rows (dh)
    const int colb = (c & 7) * 16;        // 8 x 16B per row
    async_ld_b128(lds_base + (unsigned)c * 16u,
                  (unsigned long long)(size_t)(gsrc + (size_t)row * T) +
                      (unsigned long long)colb);
  }
}

// ---------------------------------------------------------------- cvt f32->bf16
__global__ __launch_bounds__(256) void cvt_bf16_kernel(
    const float* __restrict__ src, bf16_t* __restrict__ dst, int n4) {
  int i = blockIdx.x * 256 + threadIdx.x;
  if (i < n4) {
    v4f v = *reinterpret_cast<const v4f*>(src + (size_t)i * 4);
    *reinterpret_cast<v4bf*>(dst + (size_t)i * 4) = __builtin_convertvector(v, v4bf);
  }
}

// ---------------------------------------------------------------- QKV projection
// One wave computes a 16(M) x 64(N) strip of Y = X @ W^T + b. The k-loop is
// unrolled 2x into a ping-pong: each fragment set is consumed and then
// reloaded IN PLACE for k+64, so there are no register-rotation moves and each
// set's loads get a full half-iteration (4 WMMAs) of latency cover.
__global__ __launch_bounds__(32) void qkv_proj_kernel(
    const bf16_t* __restrict__ Xbf,
    const bf16_t* __restrict__ Wq, const bf16_t* __restrict__ Wk,
    const bf16_t* __restrict__ Wv,
    const float* __restrict__ bq, const float* __restrict__ bk,
    const float* __restrict__ bv,
    bf16_t* __restrict__ Qb, bf16_t* __restrict__ Kb, bf16_t* __restrict__ Vtb) {
  const int m0 = blockIdx.x * 16;
  const int n0 = blockIdx.y * 64;
  const int p  = blockIdx.z;
  const bf16_t* W    = (p == 0) ? Wq : (p == 1) ? Wk : Wv;
  const float*  bias = (p == 0) ? bq : (p == 1) ? bk : bv;
  const float   scale = (p == 0) ? 0.125f : 1.0f;  // DH^-0.5 folded into Q

  const bf16_t* Xp = Xbf + (size_t)m0 * D;
  v8f acc[4] = {{}, {}, {}, {}};

  v16bf a0 = load_frag(Xp + 0, D);
  v16bf b0[4];
#pragma unroll
  for (int nt = 0; nt < 4; ++nt)
    b0[nt] = load_frag(W + (size_t)(n0 + nt * 16) * D + 0, D);
  v16bf a1 = load_frag(Xp + 32, D);
  v16bf b1[4];
#pragma unroll
  for (int nt = 0; nt < 4; ++nt)
    b1[nt] = load_frag(W + (size_t)(n0 + nt * 16) * D + 32, D);

#pragma unroll 1
  for (int k0 = 64; k0 < D; k0 += 64) {
    __builtin_prefetch(Xp + k0 + 64, 0, 0);
#pragma unroll
    for (int nt = 0; nt < 4; ++nt) acc[nt] = wmma_bf16(a0, b0[nt], acc[nt]);
    a0 = load_frag(Xp + k0, D);
#pragma unroll
    for (int nt = 0; nt < 4; ++nt)
      b0[nt] = load_frag(W + (size_t)(n0 + nt * 16) * D + k0, D);
#pragma unroll
    for (int nt = 0; nt < 4; ++nt) acc[nt] = wmma_bf16(a1, b1[nt], acc[nt]);
    a1 = load_frag(Xp + k0 + 32, D);
#pragma unroll
    for (int nt = 0; nt < 4; ++nt)
      b1[nt] = load_frag(W + (size_t)(n0 + nt * 16) * D + k0 + 32, D);
  }
#pragma unroll
  for (int nt = 0; nt < 4; ++nt) acc[nt] = wmma_bf16(a0, b0[nt], acc[nt]);
#pragma unroll
  for (int nt = 0; nt < 4; ++nt) acc[nt] = wmma_bf16(a1, b1[nt], acc[nt]);

  const int lane = threadIdx.x & 31;
  const int nlo  = lane & 15;
  const int mofs = (lane >> 4) << 3;  // C layout: lanes>=16 hold M = r+8
#pragma unroll
  for (int nt = 0; nt < 4; ++nt) {
    const int n  = n0 + nt * 16 + nlo;
    const float bn = bias[n];
    const int h = n >> 6, dh = n & 63;
#pragma unroll
    for (int r = 0; r < 8; ++r) {
      const int m  = m0 + r + mofs;
      const int b  = m >> 11;          // T = 2048
      const int t  = m & (T - 1);
      const int bh = b * H + h;
      float y = clamp_nan((acc[nt][r] + bn) * scale);
      bf16_t yb = (bf16_t)y;
      if (p == 0)      Qb[((size_t)bh * T + t) * DH + dh] = yb;  // [bh,t,dh]
      else if (p == 1) Kb[((size_t)bh * T + t) * DH + dh] = yb;  // [bh,t,dh]
      else             Vtb[((size_t)bh * DH + dh) * T + t] = yb; // [bh,dh,t]
    }
  }
}

// ---------------------------------------------------------------- attention core
// One block = 4 waves = 64 query rows of one (b,h). K tiles are DMA'd into LDS
// by the Tensor Data Mover (wave 0 issues tensor_load_to_lds, TENSORcnt);
// V tiles stream in via global_load_async_to_lds_b128 (ASYNCcnt). Both are
// double-buffered so the next tile is in flight while WMMAs consume the
// current one. Clamped scores are <= 1 so no online max; clamp(P) is provably
// inactive -> single pass O = (sum exp(S) V) / (sum exp(S)); row sums via WMMA
// against an all-ones B fragment (lands in the C/D layout).
__global__ __launch_bounds__(128) void attn_kernel(
    const bf16_t* __restrict__ Qb, const bf16_t* __restrict__ Kb,
    const bf16_t* __restrict__ Vtb, bf16_t* __restrict__ AOb) {
  const int wave = threadIdx.x >> 5;
  const int lane = threadIdx.x & 31;
  const int qt = blockIdx.x * 64 + wave * 16;  // query tile start
  const int bh = blockIdx.y;

  const bf16_t* Qh = Qb  + (size_t)bh * T * DH;
  const bf16_t* Kh = Kb  + (size_t)bh * T * DH;
  const bf16_t* Vh = Vtb + (size_t)bh * DH * T;

  __shared__ __align__(16) bf16_t kt[2][64 * 64];  // [buf][key][dh]  8 KB each
  __shared__ __align__(16) bf16_t vt[2][64 * 64];  // [buf][dh][key]  8 KB each
  __shared__ __align__(16) bf16_t pl[4][16 * 64];  // per-wave exp(S) bounce

  const unsigned ktoff[2] = {(unsigned)(size_t)&kt[0][0], (unsigned)(size_t)&kt[1][0]};
  const unsigned vtoff[2] = {(unsigned)(size_t)&vt[0][0], (unsigned)(size_t)&vt[1][0]};
  bf16_t* plw = &pl[wave][0];

  const int nlo  = lane & 15;
  const int mofs = (lane >> 4) << 3;

  v16bf qa0 = load_frag(Qh + (size_t)qt * DH +  0, DH);
  v16bf qa1 = load_frag(Qh + (size_t)qt * DH + 32, DH);

  v16bf ones;
#pragma unroll
  for (int i = 0; i < 16; ++i) ones[i] = (bf16_t)1.0f;

  v8f o[4] = {{}, {}, {}, {}};
  v8f rs = {};

  constexpr int NB = T / 64;  // 32 key blocks

  // prologue: stage buffer 0 (V: 4 async ops/wave; K: 1 TDM op on wave 0)
  stage_v_async(Vh, vtoff[0]);
  if (wave == 0) tdm_load_k64x64(ktoff[0], Kh);

  for (int i = 0; i < NB; ++i) {
    const int cur = i & 1;
    const int s0n = (i + 1) * 64;
    if (i + 1 < NB) {
      stage_v_async(Vh + s0n, vtoff[cur ^ 1]);
      if (wave == 0) tdm_load_k64x64(ktoff[cur ^ 1], Kh + (size_t)s0n * DH);
      wait_async_le4();                          // cur V retired, next in flight
      if (wave == 0) __builtin_amdgcn_s_wait_tensorcnt((short)1);
    } else {
      wait_async_0();
      if (wave == 0) __builtin_amdgcn_s_wait_tensorcnt((short)0);
    }
    __syncthreads();   // cross-wave visibility of staged tiles

    const bf16_t* kbuf = &kt[cur][0];
    const bf16_t* vbuf = &vt[cur][0];

#pragma unroll
    for (int nt = 0; nt < 4; ++nt) {
      v16bf kb0 = load_frag(kbuf + (nt * 16) * 64 +  0, 64);
      v16bf kb1 = load_frag(kbuf + (nt * 16) * 64 + 32, 64);
      v8f s = {};
      s = wmma_bf16(qa0, kb0, s);
      s = wmma_bf16(qa1, kb1, s);
#pragma unroll
      for (int r = 0; r < 8; ++r) {
        float e = __expf(clampf(s[r]));  // clamped scores: exp in (1, e]
        plw[(r + mofs) * 64 + nt * 16 + nlo] = (bf16_t)e;
      }
    }
    v16bf pa0 = load_frag(plw +  0, 64);   // ds re-layout C->A of exp(S)
    v16bf pa1 = load_frag(plw + 32, 64);
    rs = wmma_bf16(pa0, ones, rs);         // row sums via WMMA
    rs = wmma_bf16(pa1, ones, rs);
#pragma unroll
    for (int nt = 0; nt < 4; ++nt) {
      v16bf vb0 = load_frag(vbuf + (nt * 16) * 64 +  0, 64);
      v16bf vb1 = load_frag(vbuf + (nt * 16) * 64 + 32, 64);
      o[nt] = wmma_bf16(pa0, vb0, o[nt]);
      o[nt] = wmma_bf16(pa1, vb1, o[nt]);
    }
    __syncthreads();  // everyone done reading before buffer is re-staged
  }

  const int b = bh / H, h = bh % H;
#pragma unroll
  for (int r = 0; r < 8; ++r) {
    const int t = qt + r + mofs;
    const float rinv = 1.0f / rs[r];       // rs replicated across N cols
#pragma unroll
    for (int nt = 0; nt < 4; ++nt) {
      const int dh = nt * 16 + nlo;
      float val = clampf(o[nt][r] * rinv);
      AOb[((size_t)b * T + t) * D + h * DH + dh] = (bf16_t)val;  // [b,t,d]
    }
  }
}

// ---------------------------------------------------------------- out projection
__global__ __launch_bounds__(32) void out_proj_kernel(
    const bf16_t* __restrict__ AOb, const bf16_t* __restrict__ Wo,
    const float* __restrict__ bo, float* __restrict__ out) {
  const int m0 = blockIdx.x * 16;
  const int n0 = blockIdx.y * 64;
  const bf16_t* Ap = AOb + (size_t)m0 * D;

  v8f acc[4] = {{}, {}, {}, {}};
  v16bf a0 = load_frag(Ap + 0, D);
  v16bf b0[4];
#pragma unroll
  for (int nt = 0; nt < 4; ++nt)
    b0[nt] = load_frag(Wo + (size_t)(n0 + nt * 16) * D + 0, D);
  v16bf a1 = load_frag(Ap + 32, D);
  v16bf b1[4];
#pragma unroll
  for (int nt = 0; nt < 4; ++nt)
    b1[nt] = load_frag(Wo + (size_t)(n0 + nt * 16) * D + 32, D);

#pragma unroll 1
  for (int k0 = 64; k0 < D; k0 += 64) {
    __builtin_prefetch(Ap + k0 + 64, 0, 0);
#pragma unroll
    for (int nt = 0; nt < 4; ++nt) acc[nt] = wmma_bf16(a0, b0[nt], acc[nt]);
    a0 = load_frag(Ap + k0, D);
#pragma unroll
    for (int nt = 0; nt < 4; ++nt)
      b0[nt] = load_frag(Wo + (size_t)(n0 + nt * 16) * D + k0, D);
#pragma unroll
    for (int nt = 0; nt < 4; ++nt) acc[nt] = wmma_bf16(a1, b1[nt], acc[nt]);
    a1 = load_frag(Ap + k0 + 32, D);
#pragma unroll
    for (int nt = 0; nt < 4; ++nt)
      b1[nt] = load_frag(Wo + (size_t)(n0 + nt * 16) * D + k0 + 32, D);
  }
#pragma unroll
  for (int nt = 0; nt < 4; ++nt) acc[nt] = wmma_bf16(a0, b0[nt], acc[nt]);
#pragma unroll
  for (int nt = 0; nt < 4; ++nt) acc[nt] = wmma_bf16(a1, b1[nt], acc[nt]);

  const int lane = threadIdx.x & 31;
  const int nlo  = lane & 15;
  const int mofs = (lane >> 4) << 3;
#pragma unroll
  for (int nt = 0; nt < 4; ++nt) {
    const int n = n0 + nt * 16 + nlo;
    const float bn = bo[n];
#pragma unroll
    for (int r = 0; r < 8; ++r) {
      const int m = m0 + r + mofs;
      out[(size_t)m * D + n] = clamp_nan(acc[nt][r] + bn);
    }
  }
}

// ---------------------------------------------------------------- launch
extern "C" void kernel_launch(void* const* d_in, const int* in_sizes, int n_in,
                              void* d_out, int out_size, void* d_ws, size_t ws_size,
                              hipStream_t stream) {
  const float* X  = (const float*)d_in[0];
  const float* Wq = (const float*)d_in[1];
  const float* bq = (const float*)d_in[2];
  const float* Wk = (const float*)d_in[3];
  const float* bk = (const float*)d_in[4];
  const float* Wv = (const float*)d_in[5];
  const float* bv = (const float*)d_in[6];
  const float* Wo = (const float*)d_in[7];
  const float* bo = (const float*)d_in[8];
  float* out = (float*)d_out;

  const size_t NX = (size_t)B * T * D;  // 4,194,304
  const size_t NW = (size_t)D * D;      // 1,048,576
  // ws layout (bf16): X | Wq Wk Wv Wo | Q | K | Vt | AO  (48 MiB total)
  bf16_t* Xbf = (bf16_t*)d_ws;
  bf16_t* Wqb = Xbf + NX;
  bf16_t* Wkb = Wqb + NW;
  bf16_t* Wvb = Wkb + NW;
  bf16_t* Wob = Wvb + NW;
  bf16_t* Qb  = Wob + NW;
  bf16_t* Kb  = Qb + NX;
  bf16_t* Vtb = Kb + NX;
  bf16_t* AOb = Vtb + NX;

  auto cvt = [&](const float* s, bf16_t* d, size_t n) {
    int n4 = (int)(n / 4);
    cvt_bf16_kernel<<<(n4 + 255) / 256, 256, 0, stream>>>(s, d, n4);
  };
  cvt(X,  Xbf, NX);
  cvt(Wq, Wqb, NW);
  cvt(Wk, Wkb, NW);
  cvt(Wv, Wvb, NW);
  cvt(Wo, Wob, NW);

  qkv_proj_kernel<<<dim3((B * T) / 16, D / 64, 3), 32, 0, stream>>>(
      Xbf, Wqb, Wkb, Wvb, bq, bk, bv, Qb, Kb, Vtb);
  attn_kernel<<<dim3(T / 64, B * H), 128, 0, stream>>>(Qb, Kb, Vtb, AOb);
  out_proj_kernel<<<dim3((B * T) / 16, D / 64), 32, 0, stream>>>(AOb, Wob, bo, out);
}
